// SSA_12094627906119
// MI455X (gfx1250) — compile-verified, matrix operand order checked by reading, not run on hardware
//
#include <hip/hip_runtime.h>
#include <hip/hip_bf16.h>

typedef __attribute__((ext_vector_type(16))) _Float16 v16h;
typedef __attribute__((ext_vector_type(8)))  float    v8f;
typedef __attribute__((ext_vector_type(8)))  int      v8i;

#define T_ 4
#define B_ 32
#define C_ 384
#define N_ 256
#define H_ 8
#define D_ 48


__device__ __forceinline__ v16h load_f16_frag(const _Float16* rowptr,
                                              int k0, int half) {
  v16h f;
#pragma unroll
  for (int g = 0; g < 8; ++g) {
    int kb = k0 + ((g < 4) ? (8 * half + 2 * g) : (16 + 8 * half + 2 * (g - 4)));
    f[2 * g]     = rowptr[kb];
    f[2 * g + 1] = rowptr[kb + 1];
  }
  return f;
}

__device__ __forceinline__ v8i load_u8_frag(const unsigned char* rowptr,
                                            int k0, int kmax, int half) {
  v8i f;
#pragma unroll
  for (int g = 0; g < 8; ++g) {
    int kb = k0 + 32 * (g >> 2) + 16 * ((g >> 1) & 1) + 8 * half + 4 * (g & 1);
    unsigned int val = 0u;
    if (kb < kmax) val = *(const unsigned int*)(rowptr + kb);
    f[g] = (int)val;
  }
  return f;
}


__global__ void k_cvt_w(const float* __restrict__ w,
                        _Float16* __restrict__ wh, _Float16* __restrict__ wl, int n) {
  int i = blockIdx.x * blockDim.x + threadIdx.x;
  if (i >= n) return;
  float v = w[i];
  _Float16 h = (_Float16)v;
  wh[i] = h;
  wl[i] = (_Float16)(v - (float)h);
}

__global__ void k_cvt_x(const float* __restrict__ x,
                        _Float16* __restrict__ xh, _Float16* __restrict__ xl) {
  int i = blockIdx.x * blockDim.x + threadIdx.x;
  if (i >= T_ * B_ * C_ * N_) return;
  int n = i % N_;
  int c = (i / N_) % C_;
  int tb = i / (N_ * C_);
  float v = x[i];
  _Float16 h = (_Float16)v;
  size_t o = ((size_t)tb * N_ + n) * C_ + c;
  xh[o] = h;
  xl[o] = (_Float16)(v - (float)h);
}

__global__ void k_gemm_bn(const _Float16* __restrict__ wh, const _Float16* __restrict__ wl,
                          const _Float16* __restrict__ xh, const _Float16* __restrict__ xl,
                          const float* __restrict__ gamma, const float* __restrict__ beta,
                          const float* __restrict__ mean,  const float* __restrict__ var,
                          const float* __restrict__ bias,
                          float* __restrict__ out) {
  __shared__ _Float16 sh_b[2][2][64][32];

  int blk = blockIdx.x;
  int mtg = blk % 3;
  int ntq = (blk / 3) & 3;
  int tb  = blk / 12;

  int t    = threadIdx.x;
  int lane = t & 31, w = t >> 5;
  int half = lane >> 4, l15 = lane & 15;
  int mt   = mtg * 8 + w;

  int r = t & 63, q = t >> 6;
  const _Float16* xh_row = xh + ((size_t)tb * N_ + ntq * 64 + r) * C_ + q * 8;
  const _Float16* xl_row = xl + ((size_t)tb * N_ + ntq * 64 + r) * C_ + q * 8;

  const _Float16* wh_row = wh + (size_t)(mt * 16 + l15) * C_;
  const _Float16* wl_row = wl + (size_t)(mt * 16 + l15) * C_;

  v8f acc[4] = {};
  for (int ks = 0; ks < C_ / 32; ++ks) {
    int k0 = ks * 32;
    int p  = ks & 1;
    *(uint4*)&sh_b[p][0][r][q * 8] = *(const uint4*)(xh_row + k0);
    *(uint4*)&sh_b[p][1][r][q * 8] = *(const uint4*)(xl_row + k0);
    v16h Ah = load_f16_frag(wh_row, k0, half);
    v16h Al = load_f16_frag(wl_row, k0, half);
    __syncthreads();
#pragma unroll
    for (int j = 0; j < 4; ++j) {
      v16h Bh = load_f16_frag(&sh_b[p][0][j * 16 + l15][0], 0, half);
      v16h Bl = load_f16_frag(&sh_b[p][1][j * 16 + l15][0], 0, half);
      acc[j] = __builtin_amdgcn_wmma_f32_16x16x32_f16(false, Ah, false, Bh,
                                                      (short)0, acc[j], false, false);
      acc[j] = __builtin_amdgcn_wmma_f32_16x16x32_f16(false, Ah, false, Bl,
                                                      (short)0, acc[j], false, false);
      acc[j] = __builtin_amdgcn_wmma_f32_16x16x32_f16(false, Al, false, Bh,
                                                      (short)0, acc[j], false, false);
    }
  }

#pragma unroll
  for (int rr = 0; rr < 8; ++rr) {
    int o = mt * 16 + rr + 8 * half;
    float inv = gamma[o] * rsqrtf(var[o] + 1e-5f);
    float add = beta[o] - mean[o] * inv;
    if (bias) add += bias[o] * inv;
#pragma unroll
    for (int j = 0; j < 4; ++j) {
      int n = ntq * 64 + j * 16 + l15;
      out[((size_t)tb * C_ + o) * N_ + n] = acc[j][rr] * inv + add;
    }
  }
}

__global__ void k_lif(const float* __restrict__ pre, float v_th,
                      unsigned char* __restrict__ s_nc,
                      unsigned char* __restrict__ s_cn,
                      _Float16* __restrict__ s_h16,
                      _Float16* __restrict__ s_l16,
                      float* __restrict__ s_f32)
{
  int i = blockIdx.x * blockDim.x + threadIdx.x;
  if (i >= B_ * C_ * N_) return;
  int n = i % N_;
  int c = (i / N_) % C_;
  int b = i / (N_ * C_);
  float v = 0.f;
#pragma unroll
  for (int t = 0; t < T_; ++t) {
    size_t cn = (((size_t)(t * B_ + b)) * C_ + c) * N_ + n;
    v = 0.5f * (v + pre[cn]);
    float s = (v >= v_th) ? 1.f : 0.f;
    v = v * (1.f - s);
    size_t nc = (((size_t)(t * B_ + b)) * N_ + n) * C_ + c;
    if (s_nc)  s_nc[nc] = (unsigned char)s;
    if (s_cn)  s_cn[cn] = (unsigned char)s;
    if (s_h16) { s_h16[nc] = (_Float16)s; s_l16[nc] = (_Float16)0.f; }
    if (s_f32) s_f32[cn] = s;
  }
}

__global__ void k_attn1(const unsigned char* __restrict__ q_nc,
                        const unsigned char* __restrict__ k_nc,
                        unsigned char* __restrict__ attn) {
  __shared__ unsigned char sh_k[N_][64];

  int tbh = blockIdx.x;
  int h  = tbh & (H_ - 1);
  int tb = tbh >> 3;
  int t  = threadIdx.x;

  {
    const unsigned char* src = k_nc + ((size_t)tb * N_ + t) * C_ + h * D_;
    uint4* dst = (uint4*)&sh_k[t][0];
    dst[0] = *(const uint4*)(src);
    dst[1] = *(const uint4*)(src + 16);
    dst[2] = *(const uint4*)(src + 32);
    dst[3] = make_uint4(0u, 0u, 0u, 0u);
  }
  __syncthreads();

  int lane = t & 31, w = t >> 5;
  int half = lane >> 4, l15 = lane & 15;
  unsigned char* abase = attn + (size_t)tbh * N_ * N_;

#pragma unroll
  for (int s = 0; s < 2; ++s) {
    int nt = w + s * 8;
    const unsigned char* qrow =
        q_nc + ((size_t)tb * N_ + nt * 16 + l15) * C_ + h * D_;
    v8i A = load_u8_frag(qrow, 0, D_, half);
    for (int mtile = 0; mtile < 16; ++mtile) {
      v8i B = load_u8_frag(&sh_k[mtile * 16 + l15][0], 0, 64, half);
      v8i acc = {};
      acc = __builtin_amdgcn_wmma_i32_16x16x64_iu8(false, A, false, B,
                                                   acc, false, false);
#pragma unroll
      for (int r = 0; r < 8; ++r) {
        int n = nt * 16 + r + 8 * half;
        int m = mtile * 16 + l15;
        abase[(size_t)n * N_ + m] = (unsigned char)acc[r];
      }
    }
  }
}

__global__ void k_attn2(const unsigned char* __restrict__ attn,
                        const unsigned char* __restrict__ v_cn,
                        float* __restrict__ y) {
  __shared__ unsigned char sh_v[D_][N_];

  int tbh = blockIdx.x;
  int h  = tbh & (H_ - 1);
  int tb = tbh >> 3;
  int t  = threadIdx.x;

#pragma unroll
  for (int it = 0; it < 3; ++it) {
    int idx = it * 256 + t;
    int i = idx >> 4, qq = idx & 15;
    const unsigned char* src =
        v_cn + ((size_t)tb * C_ + h * D_ + i) * N_ + qq * 16;
    *(uint4*)&sh_v[i][qq * 16] = *(const uint4*)src;
  }
  __syncthreads();

  int lane = t & 31, w = t >> 5;
  int half = lane >> 4, l15 = lane & 15;

#pragma unroll
  for (int s = 0; s < 2; ++s) {
    int nt = w + s * 8;
    const unsigned char* arow =
        attn + ((size_t)tbh * N_ + nt * 16 + l15) * N_;
    v8i acc[3] = {};
    for (int k0 = 0; k0 < N_; k0 += 64) {
      v8i A = load_u8_frag(arow, k0, N_, half);
#pragma unroll
      for (int dt = 0; dt < 3; ++dt) {
        v8i B = load_u8_frag(&sh_v[dt * 16 + l15][0], k0, N_, half);
        acc[dt] = __builtin_amdgcn_wmma_i32_16x16x64_iu8(false, A, false, B,
                                                         acc[dt], false, false);
      }
    }
#pragma unroll
    for (int dt = 0; dt < 3; ++dt) {
#pragma unroll
      for (int r = 0; r < 8; ++r) {
        int n = nt * 16 + r + 8 * half;
        int c = h * D_ + dt * 16 + l15;
        y[((size_t)tb * C_ + c) * N_ + n] = 0.25f * (float)acc[dt][r];
      }
    }
  }
}

extern "C" void kernel_launch(void* const* d_in, const int* in_sizes, int n_in,
                              void* d_out, int out_size, void* d_ws, size_t ws_size,
                              hipStream_t stream) {
  (void)in_sizes; (void)n_in; (void)out_size; (void)ws_size;

  const float* x = (const float*)d_in[0];
  const float* w_[4]  = {(const float*)d_in[1],  (const float*)d_in[6],
                         (const float*)d_in[11], (const float*)d_in[16]};
  const float* gm_[4] = {(const float*)d_in[2],  (const float*)d_in[7],
                         (const float*)d_in[12], (const float*)d_in[17]};
  const float* bt_[4] = {(const float*)d_in[3],  (const float*)d_in[8],
                         (const float*)d_in[13], (const float*)d_in[18]};
  const float* mn_[4] = {(const float*)d_in[4],  (const float*)d_in[9],
                         (const float*)d_in[14], (const float*)d_in[19]};
  const float* vr_[4] = {(const float*)d_in[5],  (const float*)d_in[10],
                         (const float*)d_in[15], (const float*)d_in[20]};
  const float* proj_b = (const float*)d_in[21];

  const size_t EL    = (size_t)T_ * B_ * C_ * N_;
  const size_t F16B  = EL * sizeof(_Float16);
  const size_t F32B  = EL * sizeof(float);
  const size_t U8B   = EL;
  const size_t WF16B = (size_t)C_ * C_ * sizeof(_Float16);
  const size_t ATTNB = (size_t)T_ * B_ * H_ * N_ * N_;

  char* ws = (char*)d_ws;
  size_t off = 0;
  _Float16* xh = (_Float16*)(ws + off); off += F16B;
  _Float16* xl = (_Float16*)(ws + off); off += F16B;
  _Float16 *wh[4], *wl[4];
  for (int p = 0; p < 4; ++p) {
    wh[p] = (_Float16*)(ws + off); off += WF16B;
    wl[p] = (_Float16*)(ws + off); off += WF16B;
  }
  float*         pre  = (float*)(ws + off);          off += F32B;
  unsigned char* q_nc = (unsigned char*)(ws + off);  off += U8B;
  unsigned char* k_nc = (unsigned char*)(ws + off);  off += U8B;
  unsigned char* v_cn = (unsigned char*)(ws + off);  off += U8B;
  unsigned char* attn = (unsigned char*)(ws + off);  off += ATTNB;
  _Float16*      sh   = (_Float16*)(ws + off);       off += F16B;
  _Float16*      sl   = (_Float16*)(ws + off);       off += F16B;

  const dim3 blk(256);
  const int WCC = C_ * C_;

  for (int p = 0; p < 4; ++p)
    k_cvt_w<<<dim3((WCC + 255) / 256), blk, 0, stream>>>(w_[p], wh[p], wl[p], WCC);
  k_cvt_x<<<dim3((unsigned)((EL + 255) / 256)), blk, 0, stream>>>(x, xh, xl);

  const dim3 gemm_grid((unsigned)(T_ * B_ * 4 * 3));
  const dim3 lif_grid((unsigned)((B_ * C_ * N_ + 255) / 256));
  const dim3 attn_grid((unsigned)(T_ * B_ * H_));

  k_gemm_bn<<<gemm_grid, blk, 0, stream>>>(wh[0], wl[0], xh, xl,
      gm_[0], bt_[0], mn_[0], vr_[0], nullptr, pre);
  k_lif<<<lif_grid, blk, 0, stream>>>(pre, 1.0f, q_nc, nullptr,
      (_Float16*)nullptr, (_Float16*)nullptr, (float*)nullptr);

  k_gemm_bn<<<gemm_grid, blk, 0, stream>>>(wh[1], wl[1], xh, xl,
      gm_[1], bt_[1], mn_[1], vr_[1], nullptr, pre);
  k_lif<<<lif_grid, blk, 0, stream>>>(pre, 1.0f, k_nc, nullptr,
      (_Float16*)nullptr, (_Float16*)nullptr, (float*)nullptr);

  k_gemm_bn<<<gemm_grid, blk, 0, stream>>>(wh[2], wl[2], xh, xl,
      gm_[2], bt_[2], mn_[2], vr_[2], nullptr, pre);
  k_lif<<<lif_grid, blk, 0, stream>>>(pre, 1.0f, nullptr, v_cn,
      (_Float16*)nullptr, (_Float16*)nullptr, (float*)nullptr);

  k_attn1<<<attn_grid, blk, 0, stream>>>(q_nc, k_nc, attn);
  k_attn2<<<attn_grid, blk, 0, stream>>>(attn, v_cn, pre);

  k_lif<<<lif_grid, blk, 0, stream>>>(pre, 0.5f, nullptr, nullptr,
      sh, sl, (float*)nullptr);

  k_gemm_bn<<<gemm_grid, blk, 0, stream>>>(wh[3], wl[3], sh, sl,
      gm_[3], bt_[3], mn_[3], vr_[3], proj_b, pre);
  k_lif<<<lif_grid, blk, 0, stream>>>(pre, 1.0f, nullptr, nullptr,
      (_Float16*)nullptr, (_Float16*)nullptr, (float*)d_out);
}
